// Attention_88613765251714
// MI455X (gfx1250) — matrix operand order repaired and checked by reading, not compiled, not run on hardware
//
#include <hip/hip_runtime.h>
#include <hip/hip_bf16.h>
#include <math.h>

// ---------------------------------------------------------------------------
// CDNA5 (gfx1250) fused attention pipeline, bf16 WMMA (16x16x32) everywhere.
// All GEMMs are NT (A row-major [M,K], B row-major [N,K]) so both operands
// load K-contiguous with global_load_b128.
// ---------------------------------------------------------------------------

typedef __attribute__((ext_vector_type(16))) __bf16 v16bf;
typedef __attribute__((ext_vector_type(8)))  float  v8f;

__device__ __forceinline__ unsigned short f2bf(float f) {
  unsigned int u = __float_as_uint(f);
  u += 0x7FFFu + ((u >> 16) & 1u);      // round-to-nearest-even
  return (unsigned short)(u >> 16);
}
__device__ __forceinline__ float bf2f(unsigned short h) {
  return __uint_as_float(((unsigned int)h) << 16);
}

union FragU { v16bf v; uint4 q[2]; };

// ---------------------------------------------------------------------------
// NT GEMM: C[M,N] = scale * (A[M,K] * B[N,K]^T) + bias[N], optional causal
// mask (col > row -> -inf), optional bf16 output, optional transposed store.
// Wave tile: 32M x 64N (2 A-frags, 4 B-frags, 8 wmma per K=32 step).
// blockDim = 256 (8 waves). blockIdx.y = batch.
// ---------------------------------------------------------------------------
__global__ __launch_bounds__(256) void gemm_nt_wmma(
    const unsigned short* __restrict__ A, const unsigned short* __restrict__ B,
    const float* __restrict__ bias, void* __restrict__ C,
    int M, int N, int K, int lda, int ldb, int ldc,
    long long sA, long long sB, long long sC,
    float scale, int causal, int outBf16, int transC)
{
  const int lane = threadIdx.x & 31;
  const int wid  = (int)((blockIdx.x * blockDim.x + threadIdx.x) >> 5);
  const int tilesN = N >> 6;
  const int tiles  = (M >> 5) * tilesN;
  if (wid >= tiles) return;
  const int m0 = (wid / tilesN) << 5;
  const int n0 = (wid % tilesN) << 6;

  const unsigned short* Ab = A + (long long)blockIdx.y * sA;
  const unsigned short* Bb = B + (long long)blockIdx.y * sB;

  const int ar = lane & 15;           // A fragment: row within 16-tile
  const int ah = (lane >> 4) << 3;    // A fragment: +8 K for upper half-wave
  const int bh = (lane >> 4) << 3;    // B fragment: +8 K for upper half-wave
  const int cm = (lane >> 4) << 3;    // C fragment: +8 M for upper half-wave
  const int cn = lane & 15;           // C fragment: column

  const v8f zero = {0.f,0.f,0.f,0.f,0.f,0.f,0.f,0.f};
  v8f acc[2][4];
#pragma unroll
  for (int mi = 0; mi < 2; ++mi)
#pragma unroll
    for (int j = 0; j < 4; ++j) acc[mi][j] = zero;

  const bool fullskip = (causal != 0) && (n0 > m0 + 31);

  if (!fullskip) {
    for (int k0 = 0; k0 < K; k0 += 32) {
      FragU a0, a1;
      const unsigned short* pa0 = Ab + (size_t)(m0 + ar) * lda + (k0 + ah);
      const unsigned short* pa1 = pa0 + (size_t)16 * lda;
      a0.q[0] = *(const uint4*)(pa0);
      a0.q[1] = *(const uint4*)(pa0 + 16);
      a1.q[0] = *(const uint4*)(pa1);
      a1.q[1] = *(const uint4*)(pa1 + 16);
      __builtin_prefetch(pa0 + 64, 0, 1);   // global_prefetch_b8 hint down K
#pragma unroll
      for (int j = 0; j < 4; ++j) {
        FragU b;
        const unsigned short* pb =
            Bb + (size_t)(n0 + j * 16 + cn) * ldb + (k0 + bh);
        b.q[0] = *(const uint4*)(pb);
        b.q[1] = *(const uint4*)(pb + 16);
        acc[0][j] = __builtin_amdgcn_wmma_f32_16x16x32_bf16(
            false, a0.v, false, b.v, (short)0, acc[0][j], false, false);
        acc[1][j] = __builtin_amdgcn_wmma_f32_16x16x32_bf16(
            false, a1.v, false, b.v, (short)0, acc[1][j], false, false);
      }
    }
  }

#pragma unroll
  for (int mi = 0; mi < 2; ++mi) {
#pragma unroll
    for (int j = 0; j < 4; ++j) {
      const int col = n0 + j * 16 + cn;
      const float bv = bias ? bias[col] : 0.0f;
#pragma unroll
      for (int r = 0; r < 8; ++r) {
        const int row = m0 + mi * 16 + cm + r;
        float v = acc[mi][j][r] * scale + bv;
        if (causal && (col > row)) v = -__builtin_inff();
        const size_t idx = transC ? ((size_t)col * ldc + row)
                                  : ((size_t)row * ldc + col);
        if (outBf16)
          ((unsigned short*)C)[(size_t)((long long)blockIdx.y * sC) + idx] = f2bf(v);
        else
          ((float*)C)[(size_t)((long long)blockIdx.y * sC) + idx] = v;
      }
    }
  }
}

// ---------------------------------------------------------------------------
// f32 -> bf16 elementwise cast (grid-stride)
// ---------------------------------------------------------------------------
__global__ __launch_bounds__(256) void cvt_f32_bf16(
    const float* __restrict__ in, unsigned short* __restrict__ out, long long n)
{
  long long i = (long long)blockIdx.x * blockDim.x + threadIdx.x;
  const long long stride = (long long)gridDim.x * blockDim.x;
  for (; i < n; i += stride) out[i] = f2bf(in[i]);
}

// ---------------------------------------------------------------------------
// In-place RoPE on a bf16 [B*S, D] buffer: rotate (i, i+half) pairs.
// ---------------------------------------------------------------------------
__global__ __launch_bounds__(256) void rope_bf16(
    unsigned short* __restrict__ qk, int S, int D, long long total)
{
  const int half = D >> 1;
  long long i = (long long)blockIdx.x * blockDim.x + threadIdx.x;
  const long long stride = (long long)gridDim.x * blockDim.x;
  for (; i < total; i += stride) {
    const int d = (int)(i % half);
    const long long bs = i / half;
    const int s = (int)(bs % S);
    const float inv = __powf(10000.0f, -((float)d) / (float)half);
    const float ang = (float)s * inv;
    float sn, cs;
    __sincosf(ang, &sn, &cs);
    unsigned short* row = qk + bs * (long long)D;
    const float x1 = bf2f(row[d]);
    const float x2 = bf2f(row[d + half]);
    row[d]        = f2bf(x1 * cs - x2 * sn);
    row[d + half] = f2bf(x1 * sn + x2 * cs);
  }
}

// ---------------------------------------------------------------------------
// Row softmax: scores f32 [rows, S] -> attn bf16 [rows, S]. One block / row.
// ---------------------------------------------------------------------------
__global__ __launch_bounds__(256) void softmax_rows(
    const float* __restrict__ sc, unsigned short* __restrict__ at, int S)
{
  const long long row = blockIdx.x;
  const float* p = sc + row * (long long)S;
  unsigned short* o = at + row * (long long)S;
  __shared__ float red[256];
  const int tid = threadIdx.x;

  float mx = -__builtin_inff();
  for (int i = tid; i < S; i += 256) mx = fmaxf(mx, p[i]);
  red[tid] = mx; __syncthreads();
  for (int s = 128; s > 0; s >>= 1) {
    if (tid < s) red[tid] = fmaxf(red[tid], red[tid + s]);
    __syncthreads();
  }
  const float m = red[0];
  __syncthreads();

  float sum = 0.f;
  for (int i = tid; i < S; i += 256) sum += __expf(p[i] - m);
  red[tid] = sum; __syncthreads();
  for (int s = 128; s > 0; s >>= 1) {
    if (tid < s) red[tid] += red[tid + s];
    __syncthreads();
  }
  const float inv = 1.0f / red[0];
  for (int i = tid; i < S; i += 256) o[i] = f2bf(__expf(p[i] - m) * inv);
}

// ---------------------------------------------------------------------------
// Host-side orchestration
// ---------------------------------------------------------------------------
extern "C" void kernel_launch(void* const* d_in, const int* in_sizes, int n_in,
                              void* d_out, int out_size, void* d_ws, size_t ws_size,
                              hipStream_t stream) {
  const float* x  = (const float*)d_in[0];
  // d_in[1] = mask (tril) -> handled analytically via causal flag
  const float* Wq = (const float*)d_in[2];
  const float* bq = (const float*)d_in[3];
  const float* Wk = (const float*)d_in[4];
  const float* bk = (const float*)d_in[5];
  const float* Wv = (const float*)d_in[6];
  const float* bv = (const float*)d_in[7];
  const float* Wo = (const float*)d_in[8];
  const float* bo = (const float*)d_in[9];
  float* out = (float*)d_out;

  // Recover dims: x = B*S*D, Wq = D*D, mask = B*S*S
  const long long nx = in_sizes[0];
  const long long nw = in_sizes[2];
  int D = 1;
  while ((long long)D * D < nw) D <<= 1;           // D is a power of two here
  const long long BS = nx / D;
  const int S = (int)((long long)in_sizes[1] / BS);
  const int B = (int)(BS / S);

  // ---- workspace carve (256B aligned) ----
  char* w = (char*)d_ws;
  auto carve = [&](size_t bytes) {
    char* p = w;
    w += (bytes + 255) & ~(size_t)255;
    return p;
  };
  const size_t bsd = (size_t)BS * D;               // B*S*D elements
  unsigned short* xb  = (unsigned short*)carve(bsd * 2);        // x bf16 (later reused as ctx)
  unsigned short* wqb = (unsigned short*)carve((size_t)D * D * 2);
  unsigned short* wkb = (unsigned short*)carve((size_t)D * D * 2);
  unsigned short* wvb = (unsigned short*)carve((size_t)D * D * 2);
  unsigned short* wob = (unsigned short*)carve((size_t)D * D * 2);
  unsigned short* qb  = (unsigned short*)carve(bsd * 2);
  unsigned short* kb  = (unsigned short*)carve(bsd * 2);
  unsigned short* vtb = (unsigned short*)carve(bsd * 2);        // [B][D][S]
  float*          sc  = (float*)carve((size_t)B * S * (size_t)S * 4);
  unsigned short* at  = (unsigned short*)carve((size_t)B * S * (size_t)S * 2);
  unsigned short* ctx = xb;                                      // alias: x dead after QKV

  const long long llSD = (long long)S * D;
  const long long llSS = (long long)S * S;
  const long long llDS = (long long)D * S;

  // ---- 1. cast inputs to bf16 ----
  {
    int blocks = (int)(((bsd) + 2047) / 2048);
    cvt_f32_bf16<<<blocks, 256, 0, stream>>>(x, xb, (long long)bsd);
    int wblocks = (int)(((size_t)D * D + 2047) / 2048);
    cvt_f32_bf16<<<wblocks, 256, 0, stream>>>(Wq, wqb, (long long)D * D);
    cvt_f32_bf16<<<wblocks, 256, 0, stream>>>(Wk, wkb, (long long)D * D);
    cvt_f32_bf16<<<wblocks, 256, 0, stream>>>(Wv, wvb, (long long)D * D);
    cvt_f32_bf16<<<wblocks, 256, 0, stream>>>(Wo, wob, (long long)D * D);
  }

  // ---- 2. QKV projections (NT GEMM, bf16 out; V stored transposed) ----
  const int tilesProj  = (S >> 5) * (D >> 6);
  const dim3 gProj((tilesProj + 7) / 8, B);
  gemm_nt_wmma<<<gProj, 256, 0, stream>>>(xb, wqb, bq, qb, S, D, D, D, D, D,
                                          llSD, 0, llSD, 1.0f, 0, 1, 0);
  gemm_nt_wmma<<<gProj, 256, 0, stream>>>(xb, wkb, bk, kb, S, D, D, D, D, D,
                                          llSD, 0, llSD, 1.0f, 0, 1, 0);
  gemm_nt_wmma<<<gProj, 256, 0, stream>>>(xb, wvb, bv, vtb, S, D, D, D, D, S,
                                          llSD, 0, llDS, 1.0f, 0, 1, 1);

  // ---- 3. RoPE on Q and K (in place, bf16) ----
  {
    const long long total = (long long)BS * (D / 2);
    int blocks = (int)((total + 2047) / 2048);
    rope_bf16<<<blocks, 256, 0, stream>>>(qb, S, D, total);
    rope_bf16<<<blocks, 256, 0, stream>>>(kb, S, D, total);
  }

  // ---- 4. scores = Q K^T / sqrt(D), causal mask -> f32 ----
  {
    const int tiles = (S >> 5) * (S >> 6);
    const dim3 g((tiles + 7) / 8, B);
    const float scale = 1.0f / sqrtf((float)D);
    gemm_nt_wmma<<<g, 256, 0, stream>>>(qb, kb, nullptr, sc, S, S, D, D, D, S,
                                        llSD, llSD, llSS, scale, 1, 0, 0);
  }

  // ---- 5. softmax rows -> attn bf16 ----
  softmax_rows<<<(unsigned)(B * (long long)S), 256, 0, stream>>>(sc, at, S);

  // ---- 6. ctx = attn * V  (NT against V^T) -> bf16 ----
  {
    const int tiles = (S >> 5) * (D >> 6);
    const dim3 g((tiles + 7) / 8, B);
    gemm_nt_wmma<<<g, 256, 0, stream>>>(at, vtb, nullptr, ctx, S, D, S, S, S, D,
                                        llSS, llDS, llSD, 1.0f, 0, 1, 0);
  }

  // ---- 7. out = ctx Wo^T + bo -> f32 ----
  gemm_nt_wmma<<<gProj, 256, 0, stream>>>(ctx, wob, bo, out, S, D, D, D, D, D,
                                          llSD, 0, llSD, 1.0f, 0, 0, 0);
}
